// GroupedQueryAttention_23270132810288
// MI455X (gfx1250) — compile-verified, matrix operand order checked by reading, not run on hardware
//
#include <hip/hip_runtime.h>
#include <hip/hip_bf16.h>

typedef __bf16 v16bf __attribute__((ext_vector_type(16)));
typedef float  v8f   __attribute__((ext_vector_type(8)));
typedef float  v4f   __attribute__((ext_vector_type(4)));
typedef unsigned int v2u __attribute__((ext_vector_type(2)));
typedef unsigned int v4u __attribute__((ext_vector_type(4)));
typedef unsigned int v8u __attribute__((ext_vector_type(8)));

#define WMMA_BF16(A, B, C) \
    __builtin_amdgcn_wmma_f32_16x16x32_bf16(false, (A), false, (B), (short)0, (C), false, false)

// ---------------------------------------------------------------------------
// helpers
// ---------------------------------------------------------------------------
__device__ __forceinline__ unsigned short f2bf(float f) {
    unsigned int u = __float_as_uint(f);
    unsigned int r = u + 0x7FFFu + ((u >> 16) & 1u);   // round-to-nearest-even
    return (unsigned short)(r >> 16);
}
__device__ __forceinline__ float bf2f(unsigned short h) {
    return __uint_as_float(((unsigned int)h) << 16);
}

// Load one 16x32 bf16 A/B fragment (K-major row of 32+ elements at p, row for
// this lane already selected).  CDNA5 16-bit layout: lanes 0-15 hold K
// {0..7,16..23}, lanes 16-31 hold K {8..15,24..31}.  Two 16B loads.
__device__ __forceinline__ v16bf frag_ld(const unsigned short* p, int lane) {
    const int off = (lane & 16) ? 8 : 0;
    union { v8u w; v4u h[2]; } u;
    u.h[0] = *(const v4u*)(p + off);
    u.h[1] = *(const v4u*)(p + off + 16);
    return __builtin_bit_cast(v16bf, u.w);
}

// CDNA5 async memory->LDS copy (16B per lane), tracked by ASYNCcnt.
// vdst VGPR = per-lane LDS byte address (low 32 bits of generic LDS pointer).
__device__ __forceinline__ void async_ld_b128(unsigned lds_addr,
                                              const unsigned short* g) {
    asm volatile("global_load_async_to_lds_b128 %0, %1, off"
                 :: "v"(lds_addr), "v"(g)
                 : "memory");
}
__device__ __forceinline__ void wait_async0() {
    asm volatile("s_wait_asynccnt 0x0" ::: "memory");
}

// ---------------------------------------------------------------------------
// fp32 -> bf16 convert, vectorized x4 (all our sizes are multiples of 4)
// ---------------------------------------------------------------------------
__global__ __launch_bounds__(256) void cvt_bf16_kernel(
    const float* __restrict__ src, unsigned int* __restrict__ dst, size_t n4) {
    size_t i = (size_t)blockIdx.x * blockDim.x + threadIdx.x;
    size_t stride = (size_t)gridDim.x * blockDim.x;
    for (; i < n4; i += stride) {
        v4f f = *(const v4f*)(src + i * 4);
        v2u o;
        o[0] = (unsigned)f2bf(f[0]) | ((unsigned)f2bf(f[1]) << 16);
        o[1] = (unsigned)f2bf(f[2]) | ((unsigned)f2bf(f[3]) << 16);
        *(v2u*)(dst + i * 2) = o;
    }
}

// ---------------------------------------------------------------------------
// GEMM: C[M,N] = A[M,K] * B[N,K]^T   (A,B bf16 K-major; C bf16 or f32)
// Block: 256 threads (8 waves), 128x128 tile, K-step 32.
// Double-buffered LDS staging via GLOBAL_LOAD_ASYNC_TO_LDS_B128 (ASYNCcnt):
// next K-tile copies overlap current tile's WMMAs; one barrier per K-step.
// ---------------------------------------------------------------------------
template <bool F32OUT>
__global__ __launch_bounds__(256) void gemm_bt_kernel(
    const unsigned short* __restrict__ A, const unsigned short* __restrict__ Bw,
    void* __restrict__ C, int M, int N, int K) {
    __shared__ unsigned short As[2][128 * 32];
    __shared__ unsigned short Bs[2][128 * 32];

    const int tid  = threadIdx.x;
    const int lane = tid & 31;
    const int wave = tid >> 5;
    const int wr   = wave >> 2;   // 0..1
    const int wc   = wave & 3;    // 0..3
    const int bm   = blockIdx.x * 128;
    const int bn   = blockIdx.y * 128;

    v8f acc[4][2];
#pragma unroll
    for (int i = 0; i < 4; ++i)
#pragma unroll
        for (int j = 0; j < 2; ++j) acc[i][j] = 0.0f;

    const int ldr = tid >> 1;          // row 0..127 staged by this thread
    const int ldh = (tid & 1) * 16;    // K half 0 / 16
    const unsigned bufB = 128u * 32u * 2u;
    const unsigned ldsA0 = (unsigned)(uintptr_t)As + (unsigned)(ldr * 32 + ldh) * 2u;
    const unsigned ldsB0 = (unsigned)(uintptr_t)Bs + (unsigned)(ldr * 32 + ldh) * 2u;

    auto stage = [&](int k0, int pb) {
        const unsigned short* ga = A  + (size_t)(bm + ldr) * K + k0 + ldh;
        const unsigned short* gb = Bw + (size_t)(bn + ldr) * K + k0 + ldh;
        const unsigned oa = ldsA0 + (unsigned)pb * bufB;
        const unsigned ob = ldsB0 + (unsigned)pb * bufB;
        async_ld_b128(oa,      ga);
        async_ld_b128(oa + 16, ga + 8);
        async_ld_b128(ob,      gb);
        async_ld_b128(ob + 16, gb + 8);
    };

    stage(0, 0);
    int p = 0;
    for (int k0 = 0; k0 < K; k0 += 32) {
        wait_async0();          // buffer p is resident for this wave
        __syncthreads();        // ... and for all waves; prior reads of p^1 done
        if (k0 + 32 < K) stage(k0 + 32, p ^ 1);   // overlap copy with compute

        v16bf af[4], bfg[2];
#pragma unroll
        for (int i = 0; i < 4; ++i)
            af[i] = frag_ld(&As[p][(wr * 64 + i * 16 + (lane & 15)) * 32], lane);
#pragma unroll
        for (int j = 0; j < 2; ++j)
            bfg[j] = frag_ld(&Bs[p][(wc * 32 + j * 16 + (lane & 15)) * 32], lane);
#pragma unroll
        for (int i = 0; i < 4; ++i)
#pragma unroll
            for (int j = 0; j < 2; ++j)
                acc[i][j] = WMMA_BF16(af[i], bfg[j], acc[i][j]);
        p ^= 1;
    }

    const int rowadd = (lane >> 4) << 3;   // C layout: lanes>=16 hold rows+8
#pragma unroll
    for (int i = 0; i < 4; ++i)
#pragma unroll
        for (int j = 0; j < 2; ++j)
#pragma unroll
            for (int v = 0; v < 8; ++v) {
                int row = bm + wr * 64 + i * 16 + rowadd + v;
                int col = bn + wc * 32 + j * 16 + (lane & 15);
                if (F32OUT)
                    ((float*)C)[(size_t)row * N + col] = acc[i][j][v];
                else
                    ((unsigned short*)C)[(size_t)row * N + col] = f2bf(acc[i][j][v]);
            }
}

// ---------------------------------------------------------------------------
// RoPE + head-major reorg.  q0 [B*T, NH*64] -> qr [B,NH,T,64]
// ---------------------------------------------------------------------------
__global__ __launch_bounds__(256) void rope_kernel(
    const unsigned short* __restrict__ src, unsigned short* __restrict__ dst,
    int T, int NH) {
    size_t idx = (size_t)blockIdx.x * blockDim.x + threadIdx.x;  // (bt*NH+h)*64+d
    int d = (int)(idx & 63);
    size_t tmp = idx >> 6;
    int h = (int)(tmp % NH);
    size_t bt = tmp / NH;
    int t = (int)(bt % T);
    int b = (int)(bt / T);

    float x1 = bf2f(src[idx]);
    float x2 = bf2f(src[idx + ((d < 32) ? 32 : -32)]);
    int j = d & 31;
    float invf = __powf(10000.0f, -(2.0f * (float)j) / 64.0f);
    float ang = (float)t * invf;
    float c = __cosf(ang), s = __sinf(ang);
    float r = (d < 32) ? (x1 * c - x2 * s) : (x1 * c + x2 * s);
    dst[(((size_t)(b * NH + h)) * T + t) * 64 + d] = f2bf(r);
}

// v0 [B*T, KV*64] -> vt [B,KV,64,T]   (transpose so P*V B-frags are contiguous)
__global__ __launch_bounds__(256) void vtrans_kernel(
    const unsigned short* __restrict__ src, unsigned short* __restrict__ dst,
    int T, int KV) {
    size_t idx = (size_t)blockIdx.x * blockDim.x + threadIdx.x;
    int d = (int)(idx & 63);
    size_t tmp = idx >> 6;
    int kv = (int)(tmp % KV);
    size_t bt = tmp / KV;
    int t = (int)(bt % T);
    int b = (int)(bt / T);
    dst[((size_t)(b * KV + kv) * 64 + d) * (size_t)T + t] = src[idx];
}

// ---------------------------------------------------------------------------
// Flash attention: block = 64 query rows (4 waves x 16 rows), causal stream
// over 64-key blocks.  K/V tiles staged ONCE per block into LDS with async
// copies (double-buffered, shared by all 4 waves); P relayout via per-wave
// LDS scratch.  yb [B*T, H*64] bf16.
// ---------------------------------------------------------------------------
__global__ __launch_bounds__(128) void flash_kernel(
    const unsigned short* __restrict__ qr, const unsigned short* __restrict__ kr,
    const unsigned short* __restrict__ vt, unsigned short* __restrict__ yb,
    int T, int H, int KV) {
    __shared__ unsigned short Ks[2][64 * 64];   // [key][d]
    __shared__ unsigned short Vs[2][64 * 64];   // [d][key]
    __shared__ unsigned short Ps[4 * 16 * 64];  // per-wave P scratch

    const int tid = threadIdx.x, lane = tid & 31, wave = tid >> 5;
    const int qb = blockIdx.x, h = blockIdx.y, b = blockIdx.z;
    const int kv = h / (H / KV);
    const int qrow0 = qb * 64 + wave * 16;
    const int rowadd = (lane >> 4) << 3;

    const unsigned short* qbase = qr + (size_t)(b * H + h) * T * 64;
    const unsigned short* kbase = kr + (size_t)(b * KV + kv) * T * 64;
    const unsigned short* vbase = vt + (size_t)(b * KV + kv) * 64 * (size_t)T;
    unsigned short* Pw = Ps + wave * 16 * 64;

    const unsigned ldsK = (unsigned)(uintptr_t)Ks;
    const unsigned ldsV = (unsigned)(uintptr_t)Vs;
    const unsigned bufB = 64u * 64u * 2u;

    // stage one 64-key K tile + V tile (16 KB) with 8 async b128 per thread
    auto stage = [&](int kbs, int pb) {
#pragma unroll
        for (int c = 0; c < 4; ++c) {
            int chunk = tid * 4 + c;        // 0..511
            int row = chunk >> 3;           // 0..63
            int col = (chunk & 7) * 8;      // bf16 element offset
            unsigned loff = (unsigned)(row * 64 + col) * 2u + (unsigned)pb * bufB;
            async_ld_b128(ldsK + loff, kbase + (size_t)(kbs + row) * 64 + col);
            async_ld_b128(ldsV + loff, vbase + (size_t)row * T + kbs + col);
        }
    };

    v16bf qf0, qf1;
    {
        const unsigned short* qp = qbase + (size_t)(qrow0 + (lane & 15)) * 64;
        qf0 = frag_ld(qp, lane);
        qf1 = frag_ld(qp + 32, lane);
    }

    v8f o[4];
#pragma unroll
    for (int nt = 0; nt < 4; ++nt) o[nt] = 0.0f;
    float mrun[8], lrun[8];
#pragma unroll
    for (int v = 0; v < 8; ++v) { mrun[v] = -3.0e38f; lrun[v] = 0.0f; }

    const int nkb = qb + 1;     // same trip count for every wave in the block
    stage(0, 0);
    int p = 0;
    for (int kb = 0; kb < nkb; ++kb) {
        const int kbs = kb * 64;
        wait_async0();          // this wave's async copies into buffer p done
        __syncthreads();        // all waves' copies done; buffer p^1 free
        if (kb + 1 < nkb) stage(kbs + 64, p ^ 1);
        if (kb + 2 < nkb)       // warm L2 two blocks ahead (global_prefetch_b8)
            __builtin_prefetch(kbase + (size_t)(kbs + 128 + (tid & 63)) * 64, 0, 1);

        // ---- S = Q K^T (4 tiles of 16 keys, K-dim 64 = 2 WMMA) ----
        v8f s[4];
#pragma unroll
        for (int nt = 0; nt < 4; ++nt) {
            const unsigned short* kp = &Ks[p][(nt * 16 + (lane & 15)) * 64];
            v16bf kf0 = frag_ld(kp, lane);
            v16bf kf1 = frag_ld(kp + 32, lane);
            v8f z = 0.0f;
            z = WMMA_BF16(qf0, kf0, z);
            z = WMMA_BF16(qf1, kf1, z);
            s[nt] = z;
        }

        // ---- scale, causal mask, row max ----
        float mx[8];
#pragma unroll
        for (int v = 0; v < 8; ++v) mx[v] = -3.0e38f;
#pragma unroll
        for (int nt = 0; nt < 4; ++nt)
#pragma unroll
            for (int v = 0; v < 8; ++v) {
                float val = s[nt][v] * 0.125f;          // 1/sqrt(64)
                int col = kbs + nt * 16 + (lane & 15);
                int row = qrow0 + rowadd + v;
                val = (col <= row) ? val : -3.0e38f;
                s[nt][v] = val;
                mx[v] = fmaxf(mx[v], val);
            }
#pragma unroll
        for (int v = 0; v < 8; ++v)
#pragma unroll
            for (int off = 8; off >= 1; off >>= 1)
                mx[v] = fmaxf(mx[v], __shfl_xor(mx[v], off, 32));

        // ---- online softmax update ----
        float rs[8];
#pragma unroll
        for (int v = 0; v < 8; ++v) {
            float mnew = fmaxf(mrun[v], mx[v]);
            float sc = __expf(mrun[v] - mnew);
            mrun[v] = mnew;
            lrun[v] *= sc;
#pragma unroll
            for (int nt = 0; nt < 4; ++nt) o[nt][v] *= sc;
            float acc = 0.0f;
#pragma unroll
            for (int nt = 0; nt < 4; ++nt) {
                float pv = __expf(s[nt][v] - mnew);
                s[nt][v] = pv;
                acc += pv;
            }
            rs[v] = acc;
        }
#pragma unroll
        for (int v = 0; v < 8; ++v) {
#pragma unroll
            for (int off = 8; off >= 1; off >>= 1)
                rs[v] += __shfl_xor(rs[v], off, 32);
            lrun[v] += rs[v];
        }

        // ---- P (C-layout) -> per-wave LDS -> A-fragment layout ----
#pragma unroll
        for (int nt = 0; nt < 4; ++nt)
#pragma unroll
            for (int v = 0; v < 8; ++v)
                Pw[(rowadd + v) * 64 + nt * 16 + (lane & 15)] = f2bf(s[nt][v]);
        asm volatile("s_wait_dscnt 0x0" ::: "memory");

        const unsigned short* pp = Pw + (size_t)(lane & 15) * 64;
        v16bf pf0 = frag_ld(pp, lane);
        v16bf pf1 = frag_ld(pp + 32, lane);

        // ---- O += P V ----
#pragma unroll
        for (int nt = 0; nt < 4; ++nt) {
            const unsigned short* vp = &Vs[p][(nt * 16 + (lane & 15)) * 64];
            v16bf vf0 = frag_ld(vp, lane);
            v16bf vf1 = frag_ld(vp + 32, lane);
            o[nt] = WMMA_BF16(pf0, vf0, o[nt]);
            o[nt] = WMMA_BF16(pf1, vf1, o[nt]);
        }
        p ^= 1;
    }

    // ---- normalize + store (back to [B,T,H*64]) ----
    float inv[8];
#pragma unroll
    for (int v = 0; v < 8; ++v) inv[v] = 1.0f / lrun[v];
#pragma unroll
    for (int nt = 0; nt < 4; ++nt)
#pragma unroll
        for (int v = 0; v < 8; ++v) {
            int t = qrow0 + rowadd + v;
            int col = h * 64 + nt * 16 + (lane & 15);
            yb[(size_t)(b * T + t) * (size_t)(H * 64) + col] =
                f2bf(o[nt][v] * inv[v]);
        }
}

// ---------------------------------------------------------------------------
// host launcher
// ---------------------------------------------------------------------------
extern "C" void kernel_launch(void* const* d_in, const int* in_sizes, int n_in,
                              void* d_out, int out_size, void* d_ws, size_t ws_size,
                              hipStream_t stream) {
    const int B = 2, T = 2048, C = 2048, H = 32, KV = 8, D = 64;
    const size_t BT = (size_t)B * T;

    const float* x  = (const float*)d_in[0];
    const float* wq = (const float*)d_in[1];
    const float* wk = (const float*)d_in[2];
    const float* wv = (const float*)d_in[3];
    const float* wo = (const float*)d_in[4];

    char* w = (char*)d_ws;
    unsigned short* xb  = (unsigned short*)w; w += BT * C * 2;
    unsigned short* wqb = (unsigned short*)w; w += (size_t)C * C * 2;
    unsigned short* wkb = (unsigned short*)w; w += (size_t)KV * D * C * 2;
    unsigned short* wvb = (unsigned short*)w; w += (size_t)KV * D * C * 2;
    unsigned short* wob = (unsigned short*)w; w += (size_t)C * C * 2;
    unsigned short* q0  = (unsigned short*)w; w += BT * C * 2;
    unsigned short* k0  = (unsigned short*)w; w += BT * KV * D * 2;
    unsigned short* v0  = (unsigned short*)w; w += BT * KV * D * 2;
    unsigned short* qr  = (unsigned short*)w; w += BT * C * 2;
    unsigned short* kr  = (unsigned short*)w; w += BT * KV * D * 2;
    unsigned short* vt  = (unsigned short*)w; w += BT * KV * D * 2;
    unsigned short* yb  = (unsigned short*)w; w += BT * C * 2;

    // 1) fp32 -> bf16 (x4 vectorized)
    cvt_bf16_kernel<<<2048, 256, 0, stream>>>(x,  (unsigned int*)xb,  BT * C / 4);
    cvt_bf16_kernel<<<2048, 256, 0, stream>>>(wq, (unsigned int*)wqb, (size_t)C * C / 4);
    cvt_bf16_kernel<<<1024, 256, 0, stream>>>(wk, (unsigned int*)wkb, (size_t)KV * D * C / 4);
    cvt_bf16_kernel<<<1024, 256, 0, stream>>>(wv, (unsigned int*)wvb, (size_t)KV * D * C / 4);
    cvt_bf16_kernel<<<2048, 256, 0, stream>>>(wo, (unsigned int*)wob, (size_t)C * C / 4);

    // 2) projections
    dim3 gq((unsigned)(BT / 128), (unsigned)(C / 128));
    dim3 gk((unsigned)(BT / 128), (unsigned)((KV * D) / 128));
    gemm_bt_kernel<false><<<gq, 256, 0, stream>>>(xb, wqb, q0, (int)BT, C, C);
    gemm_bt_kernel<false><<<gk, 256, 0, stream>>>(xb, wkb, k0, (int)BT, KV * D, C);
    gemm_bt_kernel<false><<<gk, 256, 0, stream>>>(xb, wvb, v0, (int)BT, KV * D, C);

    // 3) RoPE + head-major reorg, V transpose
    rope_kernel<<<(unsigned)(BT * C / 256), 256, 0, stream>>>(q0, qr, T, H);
    rope_kernel<<<(unsigned)(BT * KV * D / 256), 256, 0, stream>>>(k0, kr, T, KV);
    vtrans_kernel<<<(unsigned)(BT * KV * D / 256), 256, 0, stream>>>(v0, vt, T, KV);

    // 4) flash attention
    flash_kernel<<<dim3(T / 64, H, B), 128, 0, stream>>>(qr, kr, vt, yb, T, H, KV);

    // 5) output projection (fp32 epilogue into d_out)
    gemm_bt_kernel<true><<<gq, 256, 0, stream>>>(yb, wob, (float*)d_out, (int)BT, C, C);
}
